// PINN_2405181686330
// MI455X (gfx1250) — matrix operand order repaired and checked by reading, not compiled, not run on hardware
//
#include <hip/hip_runtime.h>

typedef __attribute__((ext_vector_type(16))) _Float16     v16h;
typedef __attribute__((ext_vector_type(8)))  float        v8f;
typedef __attribute__((ext_vector_type(4)))  unsigned int u32x4;
typedef __attribute__((ext_vector_type(4)))  float        f32x4;

#define HID   100
#define PADH  128
#define BPTS  32            // points per block
#define NCOLS (4*BPTS)      // 128 cols: [h | h_x | h_t | h_xx]
#define CST   136           // column stride in halves (128 + 8 pad: conflict-free b128)

// d_ws layout (bytes)
#define WS_WH   0                        // _Float16 [3][128][128] padded hidden weights
#define WS_BH   (WS_WH + 3*PADH*PADH*2)  // float    [3][128]      padded hidden biases
#define WS_W5   (WS_BH + 3*PADH*4)       // float    [2][128]      padded W5
#define WS_W1   (WS_W5 + 2*PADH*4)       // float    [128][2]      padded W1
#define WS_B1   (WS_W1 + 2*PADH*4)       // float    [128]         padded b1

union H8  { _Float16 h[8];   u32x4 u;    };
union H16 { _Float16 h[16];  u32x4 u[2]; };
union B16 { v16h     v;      u32x4 u[2]; };

// Hardware transcendental tanh (V_TANH_F32) when available; activations are
// rounded to f16 immediately, so the approximate trans op loses nothing.
__device__ __forceinline__ float fast_tanh(float v) {
#if defined(__has_builtin)
# if __has_builtin(__builtin_amdgcn_tanhf)
    return __builtin_amdgcn_tanhf(v);
# else
    return tanhf(v);
# endif
#else
    return tanhf(v);
#endif
}

// ---------------- prep: zero-padded weight copies (runs once per launch) ----------------
__global__ void pinn_prep_kernel(const float* __restrict__ W1, const float* __restrict__ b1,
                                 const float* __restrict__ W2, const float* __restrict__ b2,
                                 const float* __restrict__ W3, const float* __restrict__ b3,
                                 const float* __restrict__ W4, const float* __restrict__ b4,
                                 const float* __restrict__ W5,
                                 _Float16* __restrict__ Whp, float* __restrict__ bhp,
                                 float* __restrict__ W5p, float* __restrict__ W1p,
                                 float* __restrict__ b1p)
{
    const int i = blockIdx.x * blockDim.x + threadIdx.x;
    const float* Ws[3] = { W2, W3, W4 };
    const float* bs[3] = { b2, b3, b4 };
    if (i < 3*PADH*PADH) {
        const int L = i / (PADH*PADH), rem = i % (PADH*PADH);
        const int r = rem / PADH, k = rem % PADH;
        const float v = (r < HID && k < HID) ? Ws[L][r*HID + k] : 0.f;
        Whp[i] = (_Float16)v;
    }
    if (i < 3*PADH) {
        const int L = i / PADH, r = i % PADH;
        bhp[i] = (r < HID) ? bs[L][r] : 0.f;
    }
    if (i < 2*PADH) {
        const int o = i / PADH, k = i % PADH;
        W5p[i] = (k < HID) ? W5[o*HID + k] : 0.f;
        const int r = i >> 1, c = i & 1;                 // W1p: [128][2]
        W1p[i] = (r < HID) ? W1[r*2 + c] : 0.f;
    }
    if (i < PADH) b1p[i] = (i < HID) ? b1[i] : 0.f;
}

// ---------------- main: fused forward + tangent pushforward + residual ----------------
__global__ __launch_bounds__(256, 1)
void pinn_schrodinger_kernel(const float* __restrict__ x,  const float* __restrict__ t,
                             const _Float16* __restrict__ Whp, const float* __restrict__ bhp,
                             const float* __restrict__ W5p, const float* __restrict__ W1p,
                             const float* __restrict__ b1p, const float* __restrict__ b5,
                             float* __restrict__ out, int N)
{
    __shared__ _Float16 Hbuf[2][NCOLS * CST];   // double-buffered activations, 4 streams

    const int tid  = threadIdx.x;
    const int lane = tid & 31;
    const int wave = tid >> 5;                  // 0..7 : owns rows [16*wave, 16*wave+16)
    const int pt0  = blockIdx.x * BPTS;

    // ---------- Layer 1: analytic 2-input layer + tangent initialization ----------
    {
        const int p  = tid >> 3;                // point in block (0..31)
        const int r0 = (tid & 7) * 16;          // 16 rows per thread
        const int pt = pt0 + p;
        const float xv = (pt < N) ? x[pt] : 0.f;
        const float tv = (pt < N) ? t[pt] : 0.f;
        H16 sh, sx, st, sxx;
        #pragma unroll
        for (int rr = 0; rr < 16; ++rr) {
            const int r = r0 + rr;
            const float w0 = W1p[2*r], w1 = W1p[2*r + 1], bb = b1p[r];
            const float z   = w0*xv + w1*tv + bb;
            const float a   = fast_tanh(z);
            const float ap  = 1.f - a*a;
            const float app = -2.f * a * ap;
            sh.h[rr]  = (_Float16)a;              // h
            sx.h[rr]  = (_Float16)(ap*w0);        // h_x
            st.h[rr]  = (_Float16)(ap*w1);        // h_t
            sxx.h[rr] = (_Float16)(app*w0*w0);    // h_xx
        }
        u32x4* d0 = (u32x4*)&Hbuf[0][(0*BPTS + p)*CST + r0];
        u32x4* d1 = (u32x4*)&Hbuf[0][(1*BPTS + p)*CST + r0];
        u32x4* d2 = (u32x4*)&Hbuf[0][(2*BPTS + p)*CST + r0];
        u32x4* d3 = (u32x4*)&Hbuf[0][(3*BPTS + p)*CST + r0];
        d0[0] = sh.u[0];  d0[1] = sh.u[1];
        d1[0] = sx.u[0];  d1[1] = sx.u[1];
        d2[0] = st.u[0];  d2[1] = st.u[1];
        d3[0] = sxx.u[0]; d3[1] = sxx.u[1];
    }
    __syncthreads();

    // ---------- Layers 2..4: WMMA GEMM over 4 tangent streams + tanh pushforward ----------
    const int m0    = wave * 16;
    const int mrow  = m0 + (lane & 15);         // A-operand global row
    const int kofsA = (lane < 16) ? 0 : 8;      // ISA 16-bit A layout K offset
    const int rofsD = (lane < 16) ? 0 : 8;      // C/D layout row offset
    const int ncol  = lane & 15;                // B/C/D column within a 16-wide tile

    int cur = 0;
    #pragma unroll
    for (int L = 0; L < 3; ++L) {
        const _Float16* WL = Whp + L*PADH*PADH;
        const float*    bL = bhp + L*PADH;

        // This wave's 16x128 slab of pre-padded f16 weights -> four A operands in VGPRs.
        v16h Areg[4];
        #pragma unroll
        for (int kk = 0; kk < 4; ++kk) {
            const int kb = kk * 32;
            B16 ao;
            ao.u[0] = *(const u32x4*)(WL + mrow*PADH + kb + kofsA);
            ao.u[1] = *(const u32x4*)(WL + mrow*PADH + kb + 16 + kofsA);
            Areg[kk] = ao.v;
        }
        // Bias for the 8 C/D rows this lane owns (pre-padded -> unconditional).
        const f32x4 bv0 = *(const f32x4*)(bL + m0 + rofsD);
        const f32x4 bv1 = *(const f32x4*)(bL + m0 + rofsD + 4);
        float bk[8] = { bv0.x, bv0.y, bv0.z, bv0.w, bv1.x, bv1.y, bv1.z, bv1.w };

        #pragma unroll
        for (int g = 0; g < 2; ++g) {           // point group: cols 0-15 / 16-31 of each stream
            v8f zero = {};
            v8f acc[4] = { zero, zero, zero, zero };

            #pragma unroll
            for (int kk = 0; kk < 4; ++kk) {
                const int krow = kk*32 + ((lane < 16) ? 0 : 8);    // ISA 16-bit B layout
                #pragma unroll
                for (int s = 0; s < 4; ++s) {   // h, h_x, h_t, h_xx share the same A operand
                    const int col = s*BPTS + g*16 + ncol;
                    B16 bo;
                    const u32x4* src = (const u32x4*)&Hbuf[cur][col*CST + krow];
                    bo.u[0] = src[0];
                    bo.u[1] = src[2];
                    acc[s] = __builtin_amdgcn_wmma_f32_16x16x32_f16(
                        false, Areg[kk], false, bo.v, (short)0, acc[s], false, false);
                }
            }
            // Each lane holds z, z_x, z_t, z_xx for the SAME (row, point) slots.
            H8 oh, ox, ot, oxx;
            #pragma unroll
            for (int v = 0; v < 8; ++v) {
                const float z   = acc[0][v] + bk[v];
                const float zx  = acc[1][v];
                const float zt  = acc[2][v];
                const float zxx = acc[3][v];
                const float a   = fast_tanh(z);
                const float ap  = 1.f - a*a;
                const float app = -2.f * a * ap;
                oh.h[v]  = (_Float16)a;
                ox.h[v]  = (_Float16)(ap*zx);
                ot.h[v]  = (_Float16)(ap*zt);
                oxx.h[v] = (_Float16)(app*zx*zx + ap*zxx);
            }
            const int nxt   = cur ^ 1;
            const int ccol  = g*16 + ncol;
            const int rbase = m0 + rofsD;
            *(u32x4*)&Hbuf[nxt][(0*BPTS + ccol)*CST + rbase] = oh.u;
            *(u32x4*)&Hbuf[nxt][(1*BPTS + ccol)*CST + rbase] = ox.u;
            *(u32x4*)&Hbuf[nxt][(2*BPTS + ccol)*CST + rbase] = ot.u;
            *(u32x4*)&Hbuf[nxt][(3*BPTS + ccol)*CST + rbase] = oxx.u;
        }
        __syncthreads();
        cur ^= 1;
    }

    // ---------- Layer 5 (2x128 padded linear) + residual: 8 lanes per point ----------
    {
        const int p  = tid >> 3;                // point (0..31)
        const int q  = tid & 7;                 // 8-lane slice within the point
        const int r0 = q * 16;                  // rows [r0, r0+16)
        const int pt = pt0 + p;

        float ur = 0.f, ui = 0.f, utr = 0.f, uti = 0.f, uxr = 0.f, uxi = 0.f;
        const _Float16* hh = &Hbuf[cur][(0*BPTS + p)*CST + r0];
        const _Float16* ht = &Hbuf[cur][(2*BPTS + p)*CST + r0];
        const _Float16* hx = &Hbuf[cur][(3*BPTS + p)*CST + r0];
        #pragma unroll
        for (int c = 0; c < 2; ++c) {           // two 8-row chunks (16B LDS reads)
            H8 vh, vt, vx;
            vh.u = *(const u32x4*)(hh + 8*c);
            vt.u = *(const u32x4*)(ht + 8*c);
            vx.u = *(const u32x4*)(hx + 8*c);
            const f32x4 w0a = *(const f32x4*)(W5p + r0 + 8*c);
            const f32x4 w0b = *(const f32x4*)(W5p + r0 + 8*c + 4);
            const f32x4 w1a = *(const f32x4*)(W5p + PADH + r0 + 8*c);
            const f32x4 w1b = *(const f32x4*)(W5p + PADH + r0 + 8*c + 4);
            const float w0[8] = { w0a.x, w0a.y, w0a.z, w0a.w, w0b.x, w0b.y, w0b.z, w0b.w };
            const float w1[8] = { w1a.x, w1a.y, w1a.z, w1a.w, w1b.x, w1b.y, w1b.z, w1b.w };
            #pragma unroll
            for (int j = 0; j < 8; ++j) {
                const float hv = (float)vh.h[j], tv = (float)vt.h[j], xv = (float)vx.h[j];
                ur  += w0[j]*hv;  ui  += w1[j]*hv;
                utr += w0[j]*tv;  uti += w1[j]*tv;
                uxr += w0[j]*xv;  uxi += w1[j]*xv;
            }
        }
        // Reduce over the 8 lanes of this point (lanes 8p..8p+7, within one wave).
        #pragma unroll
        for (int m = 1; m < 8; m <<= 1) {
            ur  += __shfl_xor(ur,  m, 32);
            ui  += __shfl_xor(ui,  m, 32);
            utr += __shfl_xor(utr, m, 32);
            uti += __shfl_xor(uti, m, 32);
            uxr += __shfl_xor(uxr, m, 32);
            uxi += __shfl_xor(uxi, m, 32);
        }
        if (q == 0 && pt < N) {
            const float urr = ur + b5[0];
            const float uii = ui + b5[1];
            const float mag2 = urr*urr + uii*uii;
            out[pt]     = -uti + 0.5f*uxr + mag2*urr;   // f_r
            out[N + pt] =  utr + 0.5f*uxi + mag2*uii;   // f_i
        }
    }
}

extern "C" void kernel_launch(void* const* d_in, const int* in_sizes, int n_in,
                              void* d_out, int out_size, void* d_ws, size_t ws_size,
                              hipStream_t stream) {
    const float* x  = (const float*)d_in[0];
    const float* t  = (const float*)d_in[1];
    const float* W1 = (const float*)d_in[2];
    const float* b1 = (const float*)d_in[3];
    const float* W2 = (const float*)d_in[4];
    const float* b2 = (const float*)d_in[5];
    const float* W3 = (const float*)d_in[6];
    const float* b3 = (const float*)d_in[7];
    const float* W4 = (const float*)d_in[8];
    const float* b4 = (const float*)d_in[9];
    const float* W5 = (const float*)d_in[10];
    const float* b5 = (const float*)d_in[11];
    float* out = (float*)d_out;
    const int N = in_sizes[0];

    char* ws = (char*)d_ws;
    _Float16* Whp = (_Float16*)(ws + WS_WH);
    float*    bhp = (float*)(ws + WS_BH);
    float*    W5p = (float*)(ws + WS_W5);
    float*    W1p = (float*)(ws + WS_W1);
    float*    b1p = (float*)(ws + WS_B1);

    const int prep_elems  = 3*PADH*PADH;
    const int prep_blocks = (prep_elems + 255) / 256;
    pinn_prep_kernel<<<prep_blocks, 256, 0, stream>>>(
        W1, b1, W2, b2, W3, b3, W4, b4, W5, Whp, bhp, W5p, W1p, b1p);

    const int blocks = (N + BPTS - 1) / BPTS;
    pinn_schrodinger_kernel<<<blocks, 256, 0, stream>>>(
        x, t, Whp, bhp, W5p, W1p, b1p, b5, out, N);
}
